// MultiHeadAttention_60601988547058
// MI455X (gfx1250) — matrix operand order repaired and checked by reading, not compiled, not run on hardware
//
#include <hip/hip_runtime.h>

// ---------------- CDNA5 WMMA types ----------------
typedef __bf16 bf16_t;
typedef __attribute__((ext_vector_type(4)))  __bf16 v4bf;
typedef __attribute__((ext_vector_type(8)))  __bf16 v8bf;
typedef __attribute__((ext_vector_type(16))) __bf16 v16bf;
typedef __attribute__((ext_vector_type(8)))  float  v8f;

#define NB  2
#define NH  8
#define NL  2048
#define ND  512
#define NDH 64

// 1/sqrt(2048)
#define RSCALE 0.022097086912079612f

__device__ __forceinline__ v8f wmma_bf16(v16bf a, v16bf b, v8f c) {
  return __builtin_amdgcn_wmma_f32_16x16x32_bf16(false, a, false, b, (short)0, c,
                                                 false, false);
}

__device__ __forceinline__ v16bf cat8(v8bf lo, v8bf hi) {
  return __builtin_shufflevector(lo, hi, 0, 1, 2, 3, 4, 5, 6, 7,
                                 8, 9, 10, 11, 12, 13, 14, 15);
}

// Async 16B global -> LDS copy (CDNA5 VGLOBAL async op, tracked on ASYNCcnt).
// LDS byte offset = low 32 bits of the generic pointer (AS3->flat addrspacecast
// places the LDS offset in addr[31:0]).
__device__ __forceinline__ void async_ld_b128(const void* g, void* l) {
  const unsigned long long ga = (unsigned long long)(uintptr_t)g;
  const unsigned la = (unsigned)(uintptr_t)l;
  asm volatile("global_load_async_to_lds_b128 %0, %1, off" ::"v"(la), "v"(ga)
               : "memory");
}
__device__ __forceinline__ void wait_async0() {
  asm volatile("s_wait_asynccnt 0x0" ::: "memory");
}

// A fragment: 16(M) x 32(K) bf16 from row-major LDS tile (stride mult of 8).
// halves 0..7 -> K = hi*8 + 0..7 ; halves 8..15 -> K = 16 + hi*8 + 0..7
__device__ __forceinline__ v16bf load_a_frag(const bf16_t* base, int stride) {
  const int lane = threadIdx.x & 31;
  const int m = lane & 15;
  const int hi = lane >> 4;
  const bf16_t* row = base + m * stride + (hi << 3);
  const v8bf lo = *(const v8bf*)(row);
  const v8bf h8 = *(const v8bf*)(row + 16);
  return cat8(lo, h8);
}

// B fragment: 32(K) x 16(N) bf16 from a TRANSPOSED LDS tile: element (K=k,N=n)
// at base[n*stride + k]. halves 0..7 -> K = hi*8 + 0..7 ; halves 8..15 -> K = 16 + hi*8 + 0..7
__device__ __forceinline__ v16bf load_bt_frag(const bf16_t* base, int stride) {
  const int lane = threadIdx.x & 31;
  const int n = lane & 15;
  const int hi = lane >> 4;
  const bf16_t* p = base + n * stride + (hi << 3);
  return cat8(*(const v8bf*)(p), *(const v8bf*)(p + 16));
}

// ---------------------------------------------------------------------------
// Kernel 1: per-head projection  Y[b,h,l,e] = sum_d X[b,l,d] * W[h,d,e]  (bf16)
// grid (L/128, H, B), block 256 (8 waves). Each wave: 16 rows x 64 cols.
// ---------------------------------------------------------------------------
__global__ __launch_bounds__(256) void proj_kernel(const float* __restrict__ X,
                                                   const float* __restrict__ W,
                                                   bf16_t* __restrict__ Y) {
  const int b = blockIdx.z, h = blockIdx.y;
  const int q0 = blockIdx.x * 128;
  const int tid = threadIdx.x;
  const int wave = tid >> 5, lane = tid & 31;

  __shared__ __align__(16) bf16_t Xs[128 * 40];  // 128 x 32 row-major
  __shared__ __align__(16) bf16_t Wt[64 * 40];   // transposed: [e][d] 64 x 32
  __shared__ __align__(16) bf16_t Os[128 * 72];  // output staging 128 x 64

  const float* Xb = X + ((size_t)b * NL + q0) * ND;
  const float* Wh = W + (size_t)h * ND * NDH;

  v8f acc[4] = {};

  for (int kk = 0; kk < ND; kk += 32) {
    __syncthreads();
    // X tile: 128x32 f32 -> bf16, float4 granularity (4096 f32 = 1024 float4)
    for (int i = tid; i < 1024; i += 256) {
      const int r = i >> 3, c4 = (i & 7) << 2;
      const float4 f = *(const float4*)(Xb + (size_t)r * ND + kk + c4);
      v4bf v;
      v[0] = (bf16_t)f.x; v[1] = (bf16_t)f.y; v[2] = (bf16_t)f.z; v[3] = (bf16_t)f.w;
      *(v4bf*)&Xs[r * 40 + c4] = v;
    }
    // W tile transposed: Wt[e][d]; coalesced global reads, b128 LDS stores
    {
      const int c = tid & 63, rg = tid >> 6;  // e-column, d-row group of 8
      v8bf t;
#pragma unroll
      for (int tt = 0; tt < 8; ++tt)
        t[tt] = (bf16_t)Wh[(size_t)(kk + rg * 8 + tt) * NDH + c];
      *(v8bf*)&Wt[c * 40 + rg * 8] = t;
    }
    __syncthreads();

    const v16bf a = load_a_frag(Xs + wave * 16 * 40, 40);
#pragma unroll
    for (int nt = 0; nt < 4; ++nt) {
      const v16bf bf = load_bt_frag(Wt + nt * 16 * 40, 40);
      acc[nt] = wmma_bf16(a, bf, acc[nt]);
    }
  }

  // Epilogue: D-frag -> LDS, then coalesced b128 global stores.
  const int n = lane & 15, hi = lane >> 4;
  __syncthreads();
#pragma unroll
  for (int nt = 0; nt < 4; ++nt) {
#pragma unroll
    for (int r = 0; r < 8; ++r) {
      const int m = wave * 16 + r + hi * 8;
      Os[m * 72 + nt * 16 + n] = (bf16_t)acc[nt][r];
    }
  }
  __syncthreads();
  bf16_t* Yb = Y + ((size_t)(b * NH + h) * NL + q0) * NDH;
  for (int i = tid; i < 1024; i += 256) {  // 128*64/8 v8bf units
    const int r = i >> 3, c8 = (i & 7) << 3;
    *(v8bf*)(Yb + (size_t)r * NDH + c8) = *(const v8bf*)&Os[r * 72 + c8];
  }
}

// ---------------------------------------------------------------------------
// Kernel 2: column softmax stats over the QUERY axis.
//   y[q,k] = (Q[q,:] . K[k,:]) * RSCALE ; m[k] = max_q y ; s[k] = sum_q e^(y-m)
// grid (L/64, H, B), block 256. WG owns 64 k-columns; streams all q (32/iter).
// ---------------------------------------------------------------------------
__global__ __launch_bounds__(256) void stats_kernel(const bf16_t* __restrict__ Qb,
                                                    const bf16_t* __restrict__ Kb,
                                                    float* __restrict__ mOut,
                                                    float* __restrict__ sOut) {
  const int b = blockIdx.z, h = blockIdx.y;
  const int kbase = blockIdx.x * 64;
  const int tid = threadIdx.x, wave = tid >> 5, lane = tid & 31;
  const int nt = wave & 3, mh = wave >> 2;

  __shared__ __align__(16) bf16_t Ks[64 * 72];  // [k][e] row-major (staged once)
  __shared__ __align__(16) bf16_t Qs[32 * 72];  // [q][e] row-major
  __shared__ float redm[4][16];
  __shared__ float reds[4][16];

  const size_t bh = (size_t)(b * NH + h);
  const bf16_t* Kh = Kb + (bh * NL + kbase) * NDH;
  const bf16_t* Qh = Qb + bh * NL * NDH;

  // K tile: async global->LDS (2 x 16B per thread)
  for (int i = tid; i < 512; i += 256) {  // 64*64/8 v8bf units
    const int r = i >> 3, c8 = (i & 7) << 3;
    async_ld_b128(Kh + (size_t)r * NDH + c8, &Ks[r * 72 + c8]);
  }

  float m_run = -3.0e38f, s_run = 0.0f;

  for (int q0 = 0; q0 < NL; q0 += 32) {
    __syncthreads();
    {
      const int i = tid;  // 32*64/8 = 256 v8bf units
      const int r = i >> 3, c8 = (i & 7) << 3;
      async_ld_b128(Qh + (size_t)(q0 + r) * NDH + c8, &Qs[r * 72 + c8]);
    }
    if (q0 + 32 < NL) __builtin_prefetch(Qh + (size_t)(q0 + 32) * NDH, 0, 1);
    wait_async0();
    __syncthreads();

    v8f s = {};
    {
      const v16bf a0 = load_a_frag(Qs + mh * 16 * 72 + 0, 72);
      const v16bf b0 = load_bt_frag(Ks + nt * 16 * 72 + 0, 72);
      s = wmma_bf16(a0, b0, s);
      const v16bf a1 = load_a_frag(Qs + mh * 16 * 72 + 32, 72);
      const v16bf b1 = load_bt_frag(Ks + nt * 16 * 72 + 32, 72);
      s = wmma_bf16(a1, b1, s);
    }

    float y[8];
#pragma unroll
    for (int r = 0; r < 8; ++r) y[r] = s[r] * RSCALE;

    float tm = y[0];
#pragma unroll
    for (int r = 1; r < 8; ++r) tm = fmaxf(tm, y[r]);
    tm = fmaxf(tm, __shfl_xor(tm, 16, 32));  // combine M=0..7 with M=8..15

    const float newm = fmaxf(m_run, tm);
    float psum = 0.0f;
#pragma unroll
    for (int r = 0; r < 8; ++r) psum += __expf(y[r] - newm);
    psum += __shfl_xor(psum, 16, 32);

    s_run = s_run * __expf(m_run - newm) + psum;
    m_run = newm;
  }

  if (mh == 1 && lane < 16) {
    redm[nt][lane] = m_run;
    reds[nt][lane] = s_run;
  }
  __syncthreads();
  if (mh == 0 && lane < 16) {
    const float m2 = redm[nt][lane], s2 = reds[nt][lane];
    const float M = fmaxf(m_run, m2);
    const float S = s_run * __expf(m_run - M) + s2 * __expf(m2 - M);
    const size_t k = bh * NL + kbase + nt * 16 + lane;
    mOut[k] = M;
    sOut[k] = S;
  }
}

// ---------------------------------------------------------------------------
// Kernel 3: out[b,q,h*64+e] = sum_k exp(y[q,k]-m[k])/s[k] * V[k,e]
// grid (L/64, H, B), block 256. WG owns 64 q-rows; streams k (32/iter).
// wave w: q-tile (w>>1), e-half (w&1). P tile round-trips LDS (D->A layout).
// ---------------------------------------------------------------------------
__global__ __launch_bounds__(256) void out_kernel(const bf16_t* __restrict__ Qb,
                                                  const bf16_t* __restrict__ Kb,
                                                  const bf16_t* __restrict__ Vb,
                                                  const float* __restrict__ mIn,
                                                  const float* __restrict__ sIn,
                                                  float* __restrict__ out) {
  const int b = blockIdx.z, h = blockIdx.y;
  const int q0 = blockIdx.x * 64;
  const int tid = threadIdx.x, wave = tid >> 5, lane = tid & 31;
  const int mt = wave >> 1, eh = wave & 1;
  const int n = lane & 15, hi = lane >> 4;

  __shared__ __align__(16) bf16_t Qs[64 * 72];     // [q][e] (staged once)
  __shared__ __align__(16) bf16_t Ks[32 * 72];     // [k][e]
  __shared__ __align__(16) bf16_t Vt[64 * 40];     // transposed: [e][k] 64 x 32
  __shared__ __align__(16) bf16_t Ps[8][16 * 40];  // per-wave P tile 16q x 32k

  const size_t bh = (size_t)(b * NH + h);
  const bf16_t* Qh = Qb + (bh * NL + q0) * NDH;
  const bf16_t* Kh = Kb + bh * NL * NDH;
  const bf16_t* Vh = Vb + bh * NL * NDH;
  const float* mv = mIn + bh * NL;
  const float* sv = sIn + bh * NL;

  // Q tile: async global->LDS (2 x 16B per thread)
  for (int i = tid; i < 512; i += 256) {  // 64*64/8 v8bf units
    const int r = i >> 3, c8 = (i & 7) << 3;
    async_ld_b128(Qh + (size_t)r * NDH + c8, &Qs[r * 72 + c8]);
  }

  v8f acc0 = {}, acc1 = {};

  for (int k0 = 0; k0 < NL; k0 += 32) {
    __syncthreads();  // protect prior-iteration Ks/Vt/Ps readers
    {
      const int i = tid;  // Ks: 32*64/8 = 256 v8bf units, async
      const int r = i >> 3, c8 = (i & 7) << 3;
      async_ld_b128(Kh + (size_t)(k0 + r) * NDH + c8, &Ks[r * 72 + c8]);
    }
    {
      // Vt transposed: [e][k]; coalesced global reads, b128 LDS stores
      const int c = tid & 63, rg = tid >> 6;  // e-column, k-row group of 8
      v8bf t;
#pragma unroll
      for (int tt = 0; tt < 8; ++tt)
        t[tt] = Vh[(size_t)(k0 + rg * 8 + tt) * NDH + c];
      *(v8bf*)&Vt[c * 40 + rg * 8] = t;
    }
    if (k0 + 32 < NL) {
      __builtin_prefetch(Kh + (size_t)(k0 + 32) * NDH, 0, 1);
      __builtin_prefetch(Vh + (size_t)(k0 + 32) * NDH, 0, 1);
    }
    wait_async0();
    __syncthreads();

    // scores: 16 q (tile mt) x 32 k (this chunk)
    v8f s0 = {}, s1 = {};
    const v16bf a0 = load_a_frag(Qs + mt * 16 * 72 + 0, 72);
    const v16bf a1 = load_a_frag(Qs + mt * 16 * 72 + 32, 72);
    const v16bf b00 = load_bt_frag(Ks + 0 * 72 + 0, 72);
    const v16bf b01 = load_bt_frag(Ks + 0 * 72 + 32, 72);
    const v16bf b10 = load_bt_frag(Ks + 16 * 72 + 0, 72);
    const v16bf b11 = load_bt_frag(Ks + 16 * 72 + 32, 72);
    s0 = wmma_bf16(a0, b00, s0);
    s0 = wmma_bf16(a1, b01, s0);
    s1 = wmma_bf16(a0, b10, s1);
    s1 = wmma_bf16(a1, b11, s1);

    // probabilities (column-k normalizers); D-layout -> LDS in A-layout
    const float m0 = mv[k0 + n];
    const float i0 = 1.0f / sv[k0 + n];
    const float m1 = mv[k0 + 16 + n];
    const float i1 = 1.0f / sv[k0 + 16 + n];
    bf16_t* P = &Ps[wave][0];
#pragma unroll
    for (int r = 0; r < 8; ++r) {
      const int row = r + hi * 8;
      P[row * 40 + n]      = (bf16_t)(__expf(s0[r] * RSCALE - m0) * i0);
      P[row * 40 + 16 + n] = (bf16_t)(__expf(s1[r] * RSCALE - m1) * i1);
    }
    __syncthreads();  // make P visible across the wave's lanes

    const v16bf ap  = load_a_frag(P, 40);
    const v16bf vb0 = load_bt_frag(Vt + (eh * 32 + 0) * 40, 40);
    const v16bf vb1 = load_bt_frag(Vt + (eh * 32 + 16) * 40, 40);
    acc0 = wmma_bf16(ap, vb0, acc0);
    acc1 = wmma_bf16(ap, vb1, acc1);
  }

  float* ob = out + ((size_t)b * NL + q0 + mt * 16) * (NH * NDH) + h * NDH + eh * 32;
#pragma unroll
  for (int r = 0; r < 8; ++r) {
    const int row = r + hi * 8;
    ob[(size_t)row * (NH * NDH) + n]      = acc0[r];
    ob[(size_t)row * (NH * NDH) + 16 + n] = acc1[r];
  }
}

// ---------------------------------------------------------------------------
extern "C" void kernel_launch(void* const* d_in, const int* in_sizes, int n_in,
                              void* d_out, int out_size, void* d_ws, size_t ws_size,
                              hipStream_t stream) {
  (void)in_sizes; (void)n_in; (void)out_size; (void)ws_size;
  const float* keys    = (const float*)d_in[0];
  const float* queries = (const float*)d_in[1];
  const float* values  = (const float*)d_in[2];
  const float* WQ      = (const float*)d_in[3];
  const float* WK      = (const float*)d_in[4];
  const float* WV      = (const float*)d_in[5];
  float* out = (float*)d_out;

  char* w = (char*)d_ws;
  const size_t projBytes = (size_t)NB * NH * NL * NDH * sizeof(bf16_t);  // 4 MiB
  bf16_t* Qb = (bf16_t*)(w);
  bf16_t* Kb = (bf16_t*)(w + projBytes);
  bf16_t* Vb = (bf16_t*)(w + 2 * projBytes);
  float*  mA = (float*)(w + 3 * projBytes);
  float*  sA = (float*)(w + 3 * projBytes + (size_t)NB * NH * NL * sizeof(float));

  const dim3 blk(256);
  proj_kernel<<<dim3(NL / 128, NH, NB), blk, 0, stream>>>(queries, WQ, Qb);
  proj_kernel<<<dim3(NL / 128, NH, NB), blk, 0, stream>>>(keys,    WK, Kb);
  proj_kernel<<<dim3(NL / 128, NH, NB), blk, 0, stream>>>(values,  WV, Vb);
  stats_kernel<<<dim3(NL / 64, NH, NB), blk, 0, stream>>>(Qb, Kb, mA, sA);
  out_kernel<<<dim3(NL / 64, NH, NB), blk, 0, stream>>>(Qb, Kb, Vb, mA, sA, out);
}
